// Model_48696339202350
// MI455X (gfx1250) — matrix operand order repaired and checked by reading, not compiled, not run on hardware
//
#include <hip/hip_runtime.h>
#include <math.h>

typedef __attribute__((ext_vector_type(16))) _Float16 v16h;
typedef __attribute__((ext_vector_type(8)))  float    v8f;

#define B_ 16
#define T_ 750
#define F_ 4096
#define C_ 20
#define K_ 93           // T_/8
#define NTILE 47        // ceil(T_/16)
#define NEG_HUGE (-3.402823466e+38f)

// ---------------------------------------------------------------------------
// Kernel A: fused  (x*mask)@W^T -> cas ,  ||x|| -> mag ,  x -> features copy.
// One wave per 16-row (t) tile; WMMA f32_16x16x32_f16, f32 accumulation.
// A frag (16-bit 16x32, ISA 7.12.2): lane L, row=L%16, hi=L/16:
//   elems 0..7  = K = kb + 16*hi + 0..7   (contiguous; this code's order, same as B)
//   elems 8..15 = K = kb + 16*hi + 8..15
// B frag (16-bit 32x16): lane L, col N=L%16:
//   elems 0..15 = K = kb + 16*hi + 0..15   (contiguous W row chunk)
// D (f32 16x16): lane L elem i -> (M = i + 8*hi, N = L%16)
// ---------------------------------------------------------------------------
__device__ inline void pack8(v16h& d, int base, float4 a, float4 b) {
    d[base + 0] = (_Float16)a.x; d[base + 1] = (_Float16)a.y;
    d[base + 2] = (_Float16)a.z; d[base + 3] = (_Float16)a.w;
    d[base + 4] = (_Float16)b.x; d[base + 5] = (_Float16)b.y;
    d[base + 6] = (_Float16)b.z; d[base + 7] = (_Float16)b.w;
}

__global__ __launch_bounds__(256)
void k_cas_gemm(const float* __restrict__ x, const float* __restrict__ W,
                const float* __restrict__ mask,
                float* __restrict__ cas, float* __restrict__ mag,
                float* __restrict__ feat)
{
    const int wave = threadIdx.x >> 5;             // 8 waves per block
    const int tile = blockIdx.x * 8 + wave;        // 0 .. B_*NTILE-1
    const int b  = tile / NTILE;
    const int t0 = (tile % NTILE) * 16;
    const int lane = threadIdx.x & 31;
    const int row  = lane & 15;
    const int hi   = lane >> 4;
    const int t = t0 + row;
    const int tclamp = (t < T_) ? t : (T_ - 1);
    const bool tvalid = (t < T_);

    const float* xrow = x    + ((size_t)b * T_ + tclamp) * F_;
    const float* mrow = mask + ((size_t)b * T_ + tclamp) * F_;
    float*       frow = feat + ((size_t)b * T_ + tclamp) * F_;

    const int cTile0 = row;                 // c in [0,16), always < C_
    const int cTile1 = 16 + row;            // c in [16,32), valid iff < C_
    const bool c1valid = (cTile1 < C_);
    const float* wr0 = W + (size_t)cTile0 * F_;
    const float* wr1 = W + (size_t)(c1valid ? cTile1 : 0) * F_;

    v8f acc0 = {}; v8f acc1 = {};
    float magsq = 0.f;

    for (int kb = 0; kb < F_; kb += 32) {
        const int ca = kb + 16 * hi;        // first 8-col chunk for A
        const int cb = kb + 16 * hi + 8;    // second 8-col chunk for A
        float4 x0 = *(const float4*)(xrow + ca);
        float4 x1 = *(const float4*)(xrow + ca + 4);
        float4 x2 = *(const float4*)(xrow + cb);
        float4 x3 = *(const float4*)(xrow + cb + 4);
        float4 m0 = *(const float4*)(mrow + ca);
        float4 m1 = *(const float4*)(mrow + ca + 4);
        float4 m2 = *(const float4*)(mrow + cb);
        float4 m3 = *(const float4*)(mrow + cb + 4);

        if (tvalid) {                        // features passthrough copy
            *(float4*)(frow + ca)     = x0;
            *(float4*)(frow + ca + 4) = x1;
            *(float4*)(frow + cb)     = x2;
            *(float4*)(frow + cb + 4) = x3;
        }
        magsq += x0.x*x0.x + x0.y*x0.y + x0.z*x0.z + x0.w*x0.w;
        magsq += x1.x*x1.x + x1.y*x1.y + x1.z*x1.z + x1.w*x1.w;
        magsq += x2.x*x2.x + x2.y*x2.y + x2.z*x2.z + x2.w*x2.w;
        magsq += x3.x*x3.x + x3.y*x3.y + x3.z*x3.z + x3.w*x3.w;

        float4 y0 = make_float4(x0.x*m0.x, x0.y*m0.y, x0.z*m0.z, x0.w*m0.w);
        float4 y1 = make_float4(x1.x*m1.x, x1.y*m1.y, x1.z*m1.z, x1.w*m1.w);
        float4 y2 = make_float4(x2.x*m2.x, x2.y*m2.y, x2.z*m2.z, x2.w*m2.w);
        float4 y3 = make_float4(x3.x*m3.x, x3.y*m3.y, x3.z*m3.z, x3.w*m3.w);

        v16h afrag;
        pack8(afrag, 0, y0, y1);
        pack8(afrag, 8, y2, y3);

        const int kw = kb + 16 * hi;        // contiguous 16 W values
        float4 w0 = *(const float4*)(wr0 + kw);
        float4 w1 = *(const float4*)(wr0 + kw + 4);
        float4 w2 = *(const float4*)(wr0 + kw + 8);
        float4 w3 = *(const float4*)(wr0 + kw + 12);
        v16h b0;
        pack8(b0, 0, w0, w1);
        pack8(b0, 8, w2, w3);

        v16h b1;
        if (c1valid) {
            float4 v0 = *(const float4*)(wr1 + kw);
            float4 v1 = *(const float4*)(wr1 + kw + 4);
            float4 v2 = *(const float4*)(wr1 + kw + 8);
            float4 v3 = *(const float4*)(wr1 + kw + 12);
            pack8(b1, 0, v0, v1);
            pack8(b1, 8, v2, v3);
        } else {
            float4 z = make_float4(0.f, 0.f, 0.f, 0.f);
            pack8(b1, 0, z, z);
            pack8(b1, 8, z, z);
        }

        acc0 = __builtin_amdgcn_wmma_f32_16x16x32_f16(
                   false, afrag, false, b0, (short)0, acc0, false, false);
        acc1 = __builtin_amdgcn_wmma_f32_16x16x32_f16(
                   false, afrag, false, b1, (short)0, acc1, false, false);
    }

    // lane r covers half the columns of row r; partner lane r^16 has the rest
    float tot = magsq + __shfl_xor(magsq, 16, 32);
    if (hi == 0 && tvalid) mag[(size_t)b * T_ + t] = sqrtf(tot);

#pragma unroll
    for (int i = 0; i < 8; ++i) {
        const int tt = t0 + i + 8 * hi;
        if (tt < T_) {
            const size_t base = ((size_t)b * T_ + tt) * C_;
            cas[base + cTile0] = acc0[i];
            if (c1valid) cas[base + cTile1] = acc1[i];
        }
    }
}

// ---------------------------------------------------------------------------
// Kernel B: per-batch stable top-93 index selection for act & bkg streams.
// ---------------------------------------------------------------------------
__global__ __launch_bounds__(256)
void k_topk_idx(const float* __restrict__ mag, const float* __restrict__ sel,
                int* __restrict__ idxA, int* __restrict__ idxB)
{
    const int b = blockIdx.x;
    const int tid = threadIdx.x;
    __shared__ float magv[T_];
    __shared__ float arr[T_];
    __shared__ float red[256];
    __shared__ int   redi[256];

    float lm = NEG_HUGE;
    for (int t = tid; t < T_; t += 256) {
        float v = mag[b * T_ + t];
        magv[t] = v;
        lm = fmaxf(lm, v);
    }
    red[tid] = lm; __syncthreads();
    for (int s = 128; s > 0; s >>= 1) {
        if (tid < s) red[tid] = fmaxf(red[tid], red[tid + s]);
        __syncthreads();
    }
    const float maxmag = red[0];
    __syncthreads();

    for (int phase = 0; phase < 2; ++phase) {
        for (int t = tid; t < T_; t += 256) {
            float s = sel[b * T_ + t];
            arr[t] = (phase == 0) ? magv[t] * s : (maxmag - magv[t]) * s;
        }
        __syncthreads();
        int* outIdx = (phase == 0) ? idxA : idxB;
        for (int j = 0; j < K_; ++j) {
            float bv = NEG_HUGE; int bi = 0x7fffffff;
            for (int t = tid; t < T_; t += 256) {
                float v = arr[t];
                if (v > bv) { bv = v; bi = t; }   // ascending scan: ties keep lowest t
            }
            red[tid] = bv; redi[tid] = bi; __syncthreads();
            for (int s = 128; s > 0; s >>= 1) {
                if (tid < s) {
                    float ov = red[tid + s]; int oi = redi[tid + s];
                    if (ov > red[tid] || (ov == red[tid] && oi < redi[tid])) {
                        red[tid] = ov; redi[tid] = oi;
                    }
                }
                __syncthreads();
            }
            if (tid == 0) { outIdx[b * K_ + j] = redi[0]; arr[redi[0]] = NEG_HUGE; }
            __syncthreads();
        }
    }
}

// ---------------------------------------------------------------------------
// Kernel C: gather feat_act / feat_bkg rows (float4 streams).
// ---------------------------------------------------------------------------
__global__ __launch_bounds__(256)
void k_gather(const float* __restrict__ x, const int* __restrict__ idxA,
              const int* __restrict__ idxB,
              float* __restrict__ outA, float* __restrict__ outB)
{
    const int g = blockIdx.x;
    const int which = (g >= B_ * K_) ? 1 : 0;
    const int rem = which ? g - B_ * K_ : g;
    const int b = rem / K_, j = rem % K_;
    const int t = which ? idxB[b * K_ + j] : idxA[b * K_ + j];
    const float4* src = (const float4*)(x + ((size_t)b * T_ + t) * F_);
    float4* dst = (float4*)((which ? outB : outA) + ((size_t)b * K_ + j) * F_);
    for (int i = threadIdx.x; i < F_ / 4; i += 256) dst[i] = src[i];
}

// ---------------------------------------------------------------------------
// Kernel D: per-(b,c) mean of top-93 cas scores over time.
// ---------------------------------------------------------------------------
__global__ __launch_bounds__(256)
void k_topk_mean(const float* __restrict__ cas, float* __restrict__ meanAct)
{
    const int bc = blockIdx.x;
    const int b = bc / C_, c = bc % C_;
    const int tid = threadIdx.x;
    __shared__ float arr[T_];
    __shared__ float red[256];
    __shared__ int   redi[256];
    __shared__ float ssum;

    for (int t = tid; t < T_; t += 256)
        arr[t] = cas[((size_t)b * T_ + t) * C_ + c];
    if (tid == 0) ssum = 0.f;
    __syncthreads();

    for (int j = 0; j < K_; ++j) {
        float bv = NEG_HUGE; int bi = 0x7fffffff;
        for (int t = tid; t < T_; t += 256) {
            float v = arr[t];
            if (v > bv) { bv = v; bi = t; }
        }
        red[tid] = bv; redi[tid] = bi; __syncthreads();
        for (int s = 128; s > 0; s >>= 1) {
            if (tid < s) {
                float ov = red[tid + s]; int oi = redi[tid + s];
                if (ov > red[tid] || (ov == red[tid] && oi < redi[tid])) {
                    red[tid] = ov; redi[tid] = oi;
                }
            }
            __syncthreads();
        }
        if (tid == 0) { ssum += red[0]; arr[redi[0]] = NEG_HUGE; }
        __syncthreads();
    }
    if (tid == 0) meanAct[b * C_ + c] = ssum / (float)K_;
}

// ---------------------------------------------------------------------------
// Kernel E: score_act / score_bkg softmax over C per batch.
// ---------------------------------------------------------------------------
__global__ __launch_bounds__(32)
void k_scores(const float* __restrict__ cas, const float* __restrict__ meanAct,
              const int* __restrict__ idxB,
              float* __restrict__ outAct, float* __restrict__ outBkg)
{
    const int b = blockIdx.x;
    const int c = threadIdx.x;
    __shared__ float sa[C_], sb[C_];
    if (c < C_) {
        float s = 0.f;
        for (int j = 0; j < K_; ++j) {
            int t = idxB[b * K_ + j];
            s += cas[((size_t)b * T_ + t) * C_ + c];
        }
        sb[c] = s / (float)K_;
        sa[c] = meanAct[b * C_ + c];
    }
    __syncthreads();
    if (c < C_) {
        float ma = NEG_HUGE, mb = NEG_HUGE;
        for (int i = 0; i < C_; ++i) { ma = fmaxf(ma, sa[i]); mb = fmaxf(mb, sb[i]); }
        float za = 0.f, zb = 0.f;
        for (int i = 0; i < C_; ++i) { za += expf(sa[i] - ma); zb += expf(sb[i] - mb); }
        outAct[b * C_ + c] = expf(sa[c] - ma) / za;
        outBkg[b * C_ + c] = expf(sb[c] - mb) / zb;
    }
}

// ---------------------------------------------------------------------------
// Kernel F: cas_softmax over C for every (b,t).
// ---------------------------------------------------------------------------
__global__ __launch_bounds__(256)
void k_cas_softmax(const float* __restrict__ cas, float* __restrict__ out)
{
    const int i = blockIdx.x * 256 + threadIdx.x;
    if (i >= B_ * T_) return;
    const float* r = cas + (size_t)i * C_;
    float m = NEG_HUGE;
    for (int c = 0; c < C_; ++c) m = fmaxf(m, r[c]);
    float e[C_]; float z = 0.f;
    for (int c = 0; c < C_; ++c) { e[c] = expf(r[c] - m); z += e[c]; }
    const float inv = 1.f / z;
    for (int c = 0; c < C_; ++c) out[(size_t)i * C_ + c] = e[c] * inv;
}

// ---------------------------------------------------------------------------
extern "C" void kernel_launch(void* const* d_in, const int* in_sizes, int n_in,
                              void* d_out, int out_size, void* d_ws, size_t ws_size,
                              hipStream_t stream)
{
    (void)in_sizes; (void)n_in; (void)out_size; (void)ws_size;
    const float* x    = (const float*)d_in[0];
    const float* Wc   = (const float*)d_in[1];
    const float* mask = (const float*)d_in[2];
    const float* sel  = (const float*)d_in[3];
    float* out = (float*)d_out;

    // output layout (flat floats, reference tuple order)
    const size_t OFF_SACT = 0;
    const size_t OFF_SBKG = OFF_SACT + (size_t)B_ * C_;
    const size_t OFF_FACT = OFF_SBKG + (size_t)B_ * C_;
    const size_t OFF_FBKG = OFF_FACT + (size_t)B_ * K_ * F_;
    const size_t OFF_FEAT = OFF_FBKG + (size_t)B_ * K_ * F_;
    const size_t OFF_CSMX = OFF_FEAT + (size_t)B_ * T_ * F_;

    // workspace layout (~1 MB)
    float* ws      = (float*)d_ws;
    float* cas     = ws;                                  // B*T*C
    float* mag     = cas + (size_t)B_ * T_ * C_;          // B*T
    int*   idxA    = (int*)(mag + (size_t)B_ * T_);       // B*K
    int*   idxB    = idxA + (size_t)B_ * K_;              // B*K
    float* meanAct = (float*)(idxB + (size_t)B_ * K_);    // B*C

    k_cas_gemm<<<dim3((B_ * NTILE) / 8), dim3(256), 0, stream>>>(
        x, Wc, mask, cas, mag, out + OFF_FEAT);
    k_topk_idx<<<dim3(B_), dim3(256), 0, stream>>>(mag, sel, idxA, idxB);
    k_gather<<<dim3(2 * B_ * K_), dim3(256), 0, stream>>>(
        x, idxA, idxB, out + OFF_FACT, out + OFF_FBKG);
    k_topk_mean<<<dim3(B_ * C_), dim3(256), 0, stream>>>(cas, meanAct);
    k_scores<<<dim3(B_), dim3(32), 0, stream>>>(
        cas, meanAct, idxB, out + OFF_SACT, out + OFF_SBKG);
    k_cas_softmax<<<dim3((B_ * T_ + 255) / 256), dim3(256), 0, stream>>>(
        cas, out + OFF_CSMX);
}
